// GAT_PPI_88098369176194
// MI455X (gfx1250) — matrix operand order repaired and checked by reading, not compiled, not run on hardware
//
#include <hip/hip_runtime.h>

// GAT (pyGAT dense, eval mode) on MI455X / gfx1250.
// N=8192 nodes, F=256, H=4 heads x 64 hid, out 121 classes (padded 128).

#define NN   8192
#define NF   256
#define ND   64
#define NH   4
#define NC   121
#define NDP  128
#define BR   32      // rows per attention block

typedef __attribute__((ext_vector_type(16))) __bf16 v16bf;
typedef __attribute__((ext_vector_type(8)))  float  v8f;

union BF16x16 { v16bf v; unsigned int u[8]; };
union F32x8   { v8f   v; float        f[8]; };

__device__ __forceinline__ unsigned short f2bf(float f) {
  unsigned int x = __float_as_uint(f);
  x += 0x7FFFu + ((x >> 16) & 1u);          // round-to-nearest-even
  return (unsigned short)(x >> 16);
}

// A-matrix (16x32 bf16) fragment from LDS, per CDNA5 ISA 16-bit A layout:
// lane L: M = L%16; VGPR v holds K = (v>=4?16:0) + (L/16)*8 + (v%4)*2 , +1
__device__ __forceinline__ v16bf load_afrag_lds(const unsigned short* base,
                                                int rowStride, int m, int hf) {
  BF16x16 a;
#pragma unroll
  for (int v = 0; v < 8; ++v) {
    int bk = ((v & 4) ? 16 : 0) + hf * 8 + (v & 3) * 2;
    a.u[v] = *(const unsigned int*)(base + m * rowStride + bk);
  }
  return a.v;
}

// B-matrix (32x16 bf16) fragment: lane L holds column N=L%16, VGPR v holds K = (v>=4?16:0) + (L/16)*8 + (v%4)*2 , +1
__device__ __forceinline__ v16bf load_bfrag_lds(const unsigned short* colBase, int stride, int hf) {
  BF16x16 b;
#pragma unroll
  for (int v = 0; v < 8; ++v) {
    int bk = ((v & 4) ? 16 : 0) + hf * 8 + (v & 3) * 2;
    b.u[v] = (unsigned int)colBase[bk * stride] | ((unsigned int)colBase[(bk + 1) * stride] << 16);
  }
  return b.v;
}

__device__ __forceinline__ v16bf load_bfrag_g(const unsigned short* p, int stride, int hf) {
  BF16x16 b;
#pragma unroll
  for (int v = 0; v < 8; ++v) {
    int bk = ((v & 4) ? 16 : 0) + hf * 8 + (v & 3) * 2;
    b.u[v] = (unsigned int)p[bk * stride] | ((unsigned int)p[(bk + 1) * stride] << 16);
  }
  return b.v;
}

__device__ __forceinline__ v8f wmma_bf16(v16bf a, v16bf b, v8f c) {
  return __builtin_amdgcn_wmma_f32_16x16x32_bf16(false, a, false, b,
                                                 (short)0, c, false, false);
}

// ---------------- layer-1 projection: Wh[h] = x @ Ws[h] ----------------
__global__ __launch_bounds__(256) void gat_proj1(
    const float* __restrict__ x, const float* __restrict__ Ws,
    float* __restrict__ Wh, unsigned short* __restrict__ Whb) {
  __shared__ unsigned short xs[32][258];   // padded vs bank conflicts
  __shared__ unsigned short ws[256][68];
  const int h  = blockIdx.y;
  const int i0 = blockIdx.x * 32;
  const int tid = threadIdx.x;

  for (int e = tid; e < 32 * 256; e += 256) {
    int r = e >> 8, c = e & 255;
    xs[r][c] = f2bf(x[(size_t)(i0 + r) * NF + c]);
  }
  const float* W = Ws + (size_t)h * NF * ND;
  for (int e = tid; e < 256 * 64; e += 256) {
    int r = e >> 6, c = e & 63;
    ws[r][c] = f2bf(W[(size_t)r * ND + c]);
  }
  __syncthreads();

  const int wave = tid >> 5, lane = tid & 31;
  const int mr = wave >> 2, nc = wave & 3;
  const int hf = lane >> 4, lm = lane & 15;

  F32x8 acc;
#pragma unroll
  for (int v = 0; v < 8; ++v) acc.f[v] = 0.f;

  for (int k = 0; k < 8; ++k) {
    v16bf a = load_afrag_lds(&xs[mr * 16][k * 32], 258, lm, hf);
    v16bf b = load_bfrag_lds(&ws[k * 32][nc * 16 + lm], 68, hf);
    acc.v = wmma_bf16(a, b, acc.v);
  }
#pragma unroll
  for (int v = 0; v < 8; ++v) {
    int rr = i0 + mr * 16 + v + 8 * hf;
    int cc = nc * 16 + lm;
    size_t idx = ((size_t)h * NN + rr) * ND + cc;
    float val = acc.f[v];
    Wh[idx]  = val;
    Whb[idx] = f2bf(val);
  }
}

// ---------------- layer-1 attention vectors f_src / f_dst ----------------
__global__ __launch_bounds__(256) void gat_f1(
    const float* __restrict__ Wh, const float* __restrict__ a_heads,
    float* __restrict__ fsrc, float* __restrict__ fdst) {
  const int h = blockIdx.y;
  const int i = blockIdx.x * 256 + threadIdx.x;
  const float* a = a_heads + (size_t)h * 2 * ND;
  const float* w = Wh + ((size_t)h * NN + i) * ND;
  float s = 0.f, t = 0.f;
#pragma unroll 4
  for (int d = 0; d < ND; ++d) { float v = w[d]; s += v * a[d]; t += v * a[ND + d]; }
  fsrc[(size_t)h * NN + i] = s;
  fdst[(size_t)h * NN + i] = t;
}

// ---------- layer-1 fused flash-GAT: 32 rows x 4 heads per block ----------
__global__ __launch_bounds__(256) void gat_attn1(
    const float* __restrict__ adj,
    const float* __restrict__ fsrc, const float* __restrict__ fdst,
    const unsigned short* __restrict__ Whb,
    float* __restrict__ hcat) {
  __shared__ float          adjt[BR][33];
  __shared__ unsigned short pt[NH][BR][34];
  __shared__ float arow[NH][BR];
  __shared__ float lrowS[NH][BR];
  __shared__ float fdstS[NH][32];

  const int tid  = threadIdx.x;
  const int i0   = blockIdx.x * BR;
  const int wave = tid >> 5, lane = tid & 31;
  const int wh = wave >> 1, ncp = wave & 1;          // head, col-pair for WMMA
  const int hf = lane >> 4, lm = lane & 15;
  // scoring: 2 threads per (head,row), each 16 columns, wave-shuffle combine
  const int sh  = tid >> 6;
  const int sr  = (tid & 63) >> 1;
  const int sc0 = (tid & 1) * 16;

  float m_loc = -3.0e38f, l_loc = 0.f;
  const float fs = fsrc[(size_t)sh * NN + i0 + sr];

  F32x8 acc[2][2];
#pragma unroll
  for (int rt = 0; rt < 2; ++rt)
#pragma unroll
    for (int v = 0; v < 8; ++v) { acc[rt][0].f[v] = 0.f; acc[rt][1].f[v] = 0.f; }

  for (int kc = 0; kc < NN / 32; ++kc) {
    const int j0 = kc * 32;
    {   // stage 32x32 adj tile + fdst chunk (shared by all heads/rows)
      int r = tid >> 3, c = (tid & 7) * 4;
      const float4 a4 = *(const float4*)&adj[(size_t)(i0 + r) * NN + j0 + c];
      adjt[r][c] = a4.x; adjt[r][c + 1] = a4.y;
      adjt[r][c + 2] = a4.z; adjt[r][c + 3] = a4.w;
      if (j0 + 32 < NN)
        __builtin_prefetch(&adj[(size_t)(i0 + r) * NN + j0 + 32 + c], 0, 0);
      if (tid < NH * 32)
        fdstS[tid >> 5][tid & 31] = fdst[(size_t)(tid >> 5) * NN + j0 + (tid & 31)];
    }
    __syncthreads();

    // pass 1: masked leaky-relu scores -> new row max (registers + shuffle)
    float mx = m_loc;
#pragma unroll
    for (int c = 0; c < 16; ++c) {
      float e = fs + fdstS[sh][sc0 + c];
      e = (e >= 0.f) ? e : 0.2f * e;
      if (adjt[sr][sc0 + c] <= 0.f) e = -9.0e15f;
      mx = fmaxf(mx, e);
    }
    mx = fmaxf(mx, __shfl_xor(mx, 1, 32));
    float alpha = __expf(m_loc - mx);
    arow[sh][sr] = alpha;                 // both halves write same value
    // pass 2: recompute, exponentiate, emit bf16 P tile
    float ps = 0.f;
#pragma unroll
    for (int c = 0; c < 16; ++c) {
      float e = fs + fdstS[sh][sc0 + c];
      e = (e >= 0.f) ? e : 0.2f * e;
      if (adjt[sr][sc0 + c] <= 0.f) e = -9.0e15f;
      float p = __expf(e - mx);
      ps += p;
      pt[sh][sr][sc0 + c] = f2bf(p);
    }
    ps += __shfl_xor(ps, 1, 32);
    l_loc = l_loc * alpha + ps;
    m_loc = mx;
    __syncthreads();

    {   // WMMA: B fragments loaded once, reused across both row-tiles
      const unsigned short* bp =
          Whb + ((size_t)wh * NN + j0) * ND + ncp * 32 + lm;
      v16bf b0 = load_bfrag_g(bp, ND, hf);
      v16bf b1 = load_bfrag_g(bp + 16, ND, hf);
#pragma unroll
      for (int rt = 0; rt < 2; ++rt) {
#pragma unroll
        for (int v = 0; v < 8; ++v) {
          float al = arow[wh][rt * 16 + v + 8 * hf];
          acc[rt][0].f[v] *= al; acc[rt][1].f[v] *= al;
        }
        v16bf a = load_afrag_lds(&pt[wh][rt * 16][0], 34, lm, hf);
        acc[rt][0].v = wmma_bf16(a, b0, acc[rt][0].v);
        acc[rt][1].v = wmma_bf16(a, b1, acc[rt][1].v);
      }
    }
    __syncthreads();
  }
  lrowS[sh][sr] = l_loc;
  __syncthreads();
#pragma unroll
  for (int rt = 0; rt < 2; ++rt)
#pragma unroll
    for (int v = 0; v < 8; ++v) {
      int m = rt * 16 + v + 8 * hf;
      float inv = 1.0f / lrowS[wh][m];
      float h0 = acc[rt][0].f[v] * inv, h1 = acc[rt][1].f[v] * inv;
      h0 = (h0 > 0.f) ? h0 : (__expf(h0) - 1.0f);     // elu (concat layer)
      h1 = (h1 > 0.f) ? h1 : (__expf(h1) - 1.0f);
      size_t row = (size_t)(i0 + m);
      int cb = wh * ND + ncp * 32 + lm;
      hcat[row * (NH * ND) + cb]      = h0;
      hcat[row * (NH * ND) + cb + 16] = h1;
    }
}

// ---------------- layer-2 projection: Wh2 = h_cat @ W_out (pad 121->128) ----------------
__global__ __launch_bounds__(256) void gat_proj2(
    const float* __restrict__ hcat, const float* __restrict__ Wout,
    float* __restrict__ Wh2, unsigned short* __restrict__ Wh2b) {
  __shared__ unsigned short xs[16][258];
  __shared__ unsigned short ws[32][132];
  const int tid = threadIdx.x;
  const int i0  = blockIdx.x * 16;
  const int wave = tid >> 5, lane = tid & 31;
  const int hf = lane >> 4, lm = lane & 15;

  for (int e = tid; e < 16 * 256; e += 256) {
    int r = e >> 8, c = e & 255;
    xs[r][c] = f2bf(hcat[(size_t)(i0 + r) * (NH * ND) + c]);
  }
  F32x8 acc;
#pragma unroll
  for (int v = 0; v < 8; ++v) acc.f[v] = 0.f;

  for (int k = 0; k < 8; ++k) {
    __syncthreads();
    for (int e = tid; e < 32 * 128; e += 256) {
      int r = e >> 7, c = e & 127;
      float w = (c < NC) ? Wout[(size_t)(k * 32 + r) * NC + c] : 0.f;
      ws[r][c] = f2bf(w);
    }
    __syncthreads();
    v16bf a = load_afrag_lds(&xs[0][k * 32], 258, lm, hf);
    v16bf b = load_bfrag_lds(&ws[0][wave * 16 + lm], 132, hf);
    acc.v = wmma_bf16(a, b, acc.v);
  }
#pragma unroll
  for (int v = 0; v < 8; ++v) {
    int m = v + 8 * hf;
    size_t idx = (size_t)(i0 + m) * NDP + wave * 16 + lm;
    Wh2[idx]  = acc.f[v];
    Wh2b[idx] = f2bf(acc.f[v]);
  }
}

// ---------------- layer-2 f_src / f_dst ----------------
__global__ __launch_bounds__(256) void gat_f2(
    const float* __restrict__ Wh2, const float* __restrict__ aout,
    float* __restrict__ fsrc, float* __restrict__ fdst) {
  const int i = blockIdx.x * 256 + threadIdx.x;
  const float* w = Wh2 + (size_t)i * NDP;
  float s = 0.f, t = 0.f;
  for (int d = 0; d < NC; ++d) { float v = w[d]; s += v * aout[d]; t += v * aout[NC + d]; }
  fsrc[i] = s; fdst[i] = t;
}

// ---------------- layer-2 fused flash-GAT, 32 rows/block -> d_out[8192x121] ----------------
__global__ __launch_bounds__(256) void gat_attn2(
    const float* __restrict__ adj,
    const float* __restrict__ fsrc, const float* __restrict__ fdst,
    const unsigned short* __restrict__ Whb,
    float* __restrict__ out) {
  __shared__ float          adjt[BR][33];
  __shared__ unsigned short pt[BR][34];
  __shared__ float arow[BR];
  __shared__ float lrowS[BR];
  __shared__ float fdstS[32];

  const int tid  = threadIdx.x;
  const int i0   = blockIdx.x * BR;
  const int wave = tid >> 5, lane = tid & 31;   // wave = output col-tile (8 x 16 = 128)
  const int hf = lane >> 4, lm = lane & 15;
  // scoring: 8 threads per row, 4 columns each, butterfly combine in-wave
  const int sr  = tid >> 3;
  const int sc0 = (tid & 7) * 4;

  float m_loc = -3.0e38f, l_loc = 0.f;
  const float fs = fsrc[i0 + sr];

  F32x8 acc[2];
#pragma unroll
  for (int rt = 0; rt < 2; ++rt)
#pragma unroll
    for (int v = 0; v < 8; ++v) acc[rt].f[v] = 0.f;

  for (int kc = 0; kc < NN / 32; ++kc) {
    const int j0 = kc * 32;
    {
      int r = tid >> 3, c = (tid & 7) * 4;
      const float4 a4 = *(const float4*)&adj[(size_t)(i0 + r) * NN + j0 + c];
      adjt[r][c] = a4.x; adjt[r][c + 1] = a4.y;
      adjt[r][c + 2] = a4.z; adjt[r][c + 3] = a4.w;
      if (j0 + 32 < NN)
        __builtin_prefetch(&adj[(size_t)(i0 + r) * NN + j0 + 32 + c], 0, 0);
      if (tid < 32) fdstS[tid] = fdst[j0 + tid];
    }
    __syncthreads();

    float mx = m_loc;
#pragma unroll
    for (int c = 0; c < 4; ++c) {
      float e = fs + fdstS[sc0 + c];
      e = (e >= 0.f) ? e : 0.2f * e;
      if (adjt[sr][sc0 + c] <= 0.f) e = -9.0e15f;
      mx = fmaxf(mx, e);
    }
#pragma unroll
    for (int msk = 1; msk < 8; msk <<= 1) mx = fmaxf(mx, __shfl_xor(mx, msk, 32));
    float alpha = __expf(m_loc - mx);
    arow[sr] = alpha;
    float ps = 0.f;
#pragma unroll
    for (int c = 0; c < 4; ++c) {
      float e = fs + fdstS[sc0 + c];
      e = (e >= 0.f) ? e : 0.2f * e;
      if (adjt[sr][sc0 + c] <= 0.f) e = -9.0e15f;
      float p = __expf(e - mx);
      ps += p;
      pt[sr][sc0 + c] = f2bf(p);
    }
#pragma unroll
    for (int msk = 1; msk < 8; msk <<= 1) ps += __shfl_xor(ps, msk, 32);
    l_loc = l_loc * alpha + ps;
    m_loc = mx;
    __syncthreads();

    {
      const unsigned short* bp = Whb + (size_t)j0 * NDP + wave * 16 + lm;
      v16bf b = load_bfrag_g(bp, NDP, hf);
#pragma unroll
      for (int rt = 0; rt < 2; ++rt) {
#pragma unroll
        for (int v = 0; v < 8; ++v) acc[rt].f[v] *= arow[rt * 16 + v + 8 * hf];
        v16bf a = load_afrag_lds(&pt[rt * 16][0], 34, lm, hf);
        acc[rt].v = wmma_bf16(a, b, acc[rt].v);
      }
    }
    __syncthreads();
  }
  if ((tid & 7) == 0) lrowS[sr] = l_loc;
  __syncthreads();
#pragma unroll
  for (int rt = 0; rt < 2; ++rt)
#pragma unroll
    for (int v = 0; v < 8; ++v) {
      int m = rt * 16 + v + 8 * hf;
      int n = wave * 16 + lm;
      if (n < NC)
        out[(size_t)(i0 + m) * NC + n] = acc[rt].f[v] / lrowS[m];  // no elu
    }
}

extern "C" void kernel_launch(void* const* d_in, const int* in_sizes, int n_in,
                              void* d_out, int out_size, void* d_ws, size_t ws_size,
                              hipStream_t stream) {
  (void)in_sizes; (void)n_in; (void)out_size; (void)ws_size;
  const float* x     = (const float*)d_in[0];
  const float* adj   = (const float*)d_in[1];
  const float* Ws    = (const float*)d_in[2];
  const float* a_h   = (const float*)d_in[3];
  const float* W_out = (const float*)d_in[4];
  const float* a_out = (const float*)d_in[5];
  float* out = (float*)d_out;

  char* p = (char*)d_ws;                               // ~28 MB total scratch
  float*          Wh    = (float*)p;          p += (size_t)NH * NN * ND * 4;
  unsigned short* Whb   = (unsigned short*)p; p += (size_t)NH * NN * ND * 2;
  float*          fsrc1 = (float*)p;          p += (size_t)NH * NN * 4;
  float*          fdst1 = (float*)p;          p += (size_t)NH * NN * 4;
  float*          hcat  = (float*)p;          p += (size_t)NN * (NH * ND) * 4;
  float*          Wh2   = (float*)p;          p += (size_t)NN * NDP * 4;
  unsigned short* Wh2b  = (unsigned short*)p; p += (size_t)NN * NDP * 2;
  float*          fsrc2 = (float*)p;          p += (size_t)NN * 4;
  float*          fdst2 = (float*)p;          p += (size_t)NN * 4;

  gat_proj1<<<dim3(NN / 32, NH), 256, 0, stream>>>(x, Ws, Wh, Whb);
  gat_f1<<<dim3(NN / 256, NH), 256, 0, stream>>>(Wh, a_h, fsrc1, fdst1);
  gat_attn1<<<dim3(NN / BR), 256, 0, stream>>>(adj, fsrc1, fdst1, Whb, hcat);
  gat_proj2<<<dim3(NN / 16), 256, 0, stream>>>(hcat, W_out, Wh2, Wh2b);
  gat_f2<<<dim3(NN / 256), 256, 0, stream>>>(Wh2, a_out, fsrc2, fdst2);
  gat_attn2<<<dim3(NN / BR), 256, 0, stream>>>(adj, fsrc2, fdst2, Wh2b, out);
}
